// AccessibilitySVIGNN_36790689858192
// MI455X (gfx1250) — matrix operand order repaired and checked by reading, not compiled, not run on hardware
//
#include <hip/hip_runtime.h>
#include <math.h>

typedef __attribute__((ext_vector_type(16))) _Float16 v16h;
typedef __attribute__((ext_vector_type(8)))  float    v8f;
typedef __attribute__((ext_vector_type(4)))  _Float16 half4_t;

#define EPS 1e-5f

union Frag16 { unsigned u[8]; v16h h; };

__device__ __forceinline__ float lrelu(float v) { return v > 0.f ? v : 0.2f * v; }

__device__ __forceinline__ void atomicMaxF(float* addr, float val) {
  // sign-split trick: nonneg floats ordered as int, negatives ordered reversed as unsigned
  int iv = __float_as_int(val);
  if (iv >= 0) atomicMax((int*)addr, iv);
  else         atomicMin((unsigned*)addr, (unsigned)iv);
}

// ---------------------------------------------------------------- utilities
__global__ void k_fill(float* p, float v, long n) {
  long i = (long)blockIdx.x * blockDim.x + threadIdx.x;
  if (i < n) p[i] = v;
}

// Pack fp32 weight [Kin,Kout] into WMMA B-fragment order (f16):
// idx = (((kc*ntiles + nt)*32) + lane)*16 + h ; K = kc*32 + (lane>=16?16:0) + h ; col = nt*16 + (lane&15)
__global__ void k_pack_w(const float* __restrict__ W, _Float16* __restrict__ out,
                         int Kin, int Kout) {
  int idx = blockIdx.x * blockDim.x + threadIdx.x;
  int ntiles = Kout >> 4;
  int total = (Kin >> 5) * ntiles * 512;
  if (idx >= total) return;
  int h    = idx & 15;
  int lane = (idx >> 4) & 31;
  int rest = idx >> 9;
  int nt   = rest % ntiles;
  int kc   = rest / ntiles;
  int K    = kc * 32 + ((lane >= 16) ? 16 : 0) + h;
  int col  = nt * 16 + (lane & 15);
  out[idx] = (_Float16)W[(size_t)K * Kout + col];
}

// ---------------------------------------------------------------- LayerNorm (row of 128)
__global__ void __launch_bounds__(256) k_layernorm(const float* __restrict__ x,
                                                   const float* __restrict__ g,
                                                   const float* __restrict__ b,
                                                   float* __restrict__ out, long nrows) {
  int lane = threadIdx.x & 31, wave = threadIdx.x >> 5;
  long r = (long)blockIdx.x * 8 + wave;
  if (r >= nrows) return;
  const float4 v = ((const float4*)(x + r * 128))[lane];
  float s = v.x + v.y + v.z + v.w;
  float q = v.x * v.x + v.y * v.y + v.z * v.z + v.w * v.w;
  for (int off = 16; off > 0; off >>= 1) { s += __shfl_xor(s, off); q += __shfl_xor(q, off); }
  float mu = s * (1.f / 128.f);
  float var = q * (1.f / 128.f) - mu * mu;
  float a = rsqrtf(var + EPS);
  int k = lane * 4;
  float4 o;
  o.x = (v.x - mu) * a * g[k + 0] + b[k + 0];
  o.y = (v.y - mu) * a * g[k + 1] + b[k + 1];
  o.z = (v.z - mu) * a * g[k + 2] + b[k + 2];
  o.w = (v.w - mu) * a * g[k + 3] + b[k + 3];
  ((float4*)(out + r * 128))[lane] = o;
}

// ---------------------------------------------------------------- WMMA GEMM
// out[nrows,Kout] = post(A'[nrows,Kin] @ Wpk + bias), A' = optional BN scale/shift + relu on A.
// 256 threads = 8 waves; 128 rows per block. Wpk pre-packed f16 B-fragments.
__global__ void __launch_bounds__(256) k_gemm_wmma(
    const float* __restrict__ A, const unsigned* __restrict__ Wpk,
    const float* __restrict__ bias, const float* __restrict__ preSS,
    int preRelu, int postRelu,
    float* __restrict__ out, long nrows, int Kin, int Kout) {
  extern __shared__ _Float16 As[];  // [128][Kin]
  const int t = threadIdx.x;
  const long rowBase = (long)blockIdx.x * 128;
  const int kin4 = Kin >> 2;

  // stage A tile -> f16 LDS with optional BN+relu pre-transform
  for (int idx = t; idx < 128 * kin4; idx += 256) {
    int r = idx / kin4, k4 = idx - r * kin4;
    long gr = rowBase + r;
    float4 v = make_float4(0.f, 0.f, 0.f, 0.f);
    if (gr < nrows) v = ((const float4*)A)[(size_t)gr * kin4 + k4];
    int k = k4 * 4;
    if (preSS) {
      v.x = v.x * preSS[k + 0] + preSS[Kin + k + 0];
      v.y = v.y * preSS[k + 1] + preSS[Kin + k + 1];
      v.z = v.z * preSS[k + 2] + preSS[Kin + k + 2];
      v.w = v.w * preSS[k + 3] + preSS[Kin + k + 3];
    }
    if (preRelu) {
      v.x = fmaxf(v.x, 0.f); v.y = fmaxf(v.y, 0.f);
      v.z = fmaxf(v.z, 0.f); v.w = fmaxf(v.w, 0.f);
    }
    half4_t hv;
    hv.x = (_Float16)v.x; hv.y = (_Float16)v.y; hv.z = (_Float16)v.z; hv.w = (_Float16)v.w;
    ((half4_t*)As)[idx] = hv;
  }
  __syncthreads();

  const int lane = t & 31, wave = t >> 5;
  const int lm = lane & 15;       // A row within wave tile / D column within ntile
  const int hi = lane >> 4;       // K-half select for A, row+8 select for D
  const int ntiles = Kout >> 4, kch = Kin >> 5;
  const unsigned* Arow = (const unsigned*)(As + (size_t)(wave * 16 + lm) * Kin);

  int pk[8];
#pragma unroll
  for (int v = 0; v < 8; v++) {
    int K0 = 2 * v + (hi ? 16 : 0);
    pk[v] = K0 >> 1;  // u32 units within 32-K chunk
  }

  for (int nt = 0; nt < ntiles; nt++) {
    v8f acc = {};
    for (int kc = 0; kc < kch; kc++) {
      Frag16 a, b;
      const int cu = kc * 16;
#pragma unroll
      for (int v = 0; v < 8; v++) a.u[v] = Arow[cu + pk[v]];
      const unsigned* bp = Wpk + ((size_t)(kc * ntiles + nt) * 32 + lane) * 8;
#pragma unroll
      for (int v = 0; v < 8; v++) b.u[v] = bp[v];
      acc = __builtin_amdgcn_wmma_f32_16x16x32_f16(false, a.h, false, b.h,
                                                   (short)0, acc, false, false);
    }
    const int col = nt * 16 + lm;
    const float bc = bias ? bias[col] : 0.f;
#pragma unroll
    for (int r = 0; r < 8; r++) {
      long gr = rowBase + wave * 16 + r + 8 * hi;
      if (gr < nrows) {
        float v = acc[r] + bc;
        if (postRelu) v = fmaxf(v, 0.f);
        out[(size_t)gr * Kout + col] = v;
      }
    }
  }
}

// ---------------------------------------------------------------- GCN pieces
__global__ void k_deg_count(float* deg, const int* __restrict__ ei, long E) {
  long e = (long)blockIdx.x * blockDim.x + threadIdx.x;
  if (e < E) atomicAdd(&deg[ei[E + e]], 1.f);
}
__global__ void k_rsqrt(float* p, long n) {
  long i = (long)blockIdx.x * blockDim.x + threadIdx.x;
  if (i < n) p[i] = rsqrtf(p[i]);
}
// self-loop term + bias: out[n,c] = bias[c] + dinv[n]^2 * xw[n,c]
__global__ void k_gcn_self(float* __restrict__ out, const float* __restrict__ xw,
                           const float* __restrict__ dinv, const float* __restrict__ bias,
                           long N, int cs) {
  long tid = (long)blockIdx.x * blockDim.x + threadIdx.x;
  int C = 1 << cs;
  if (tid >= (N << cs)) return;
  long n = tid >> cs; int c = tid & (C - 1);
  float di = dinv[n];
  out[tid] = bias[c] + di * di * xw[tid];
}
// per-edge scatter: out[d,:] += xw[s,:] * dinv[s]*dinv[d]
__global__ void k_gcn_edge(float* __restrict__ out, const float* __restrict__ xw,
                           const float* __restrict__ dinv, const int* __restrict__ ei,
                           long E, int cs /*log2 C*/, int gsh /*log2 (C/4)*/) {
  long tid = (long)blockIdx.x * blockDim.x + threadIdx.x;
  if (tid >= (E << gsh)) return;
  long e = tid >> gsh; int g = tid & ((1 << gsh) - 1);
  int s = ei[e], d = ei[E + e];
  float w = dinv[s] * dinv[d];
  const int C = 1 << cs;
  float4 v = ((const float4*)(xw + (size_t)s * C))[g];
  float* o = out + (size_t)d * C + g * 4;
  atomicAdd(o + 0, v.x * w); atomicAdd(o + 1, v.y * w);
  atomicAdd(o + 2, v.z * w); atomicAdd(o + 3, v.w * w);
}

// ---------------------------------------------------------------- BatchNorm
__global__ void __launch_bounds__(256) k_bn_stats(const float* __restrict__ x,
                                                  float* __restrict__ sums, long nrows, int cs) {
  int C = 1 << cs;
  int t = threadIdx.x, c = t & (C - 1), rsub = t >> cs;
  int rowsPer = 256 >> cs;
  float s = 0.f, q = 0.f;
  for (long r = (long)blockIdx.x * rowsPer + rsub; r < nrows; r += (long)gridDim.x * rowsPer) {
    float v = x[(size_t)r * C + c];
    s += v; q += v * v;
  }
  atomicAdd(&sums[c], s);
  atomicAdd(&sums[C + c], q);
}
__global__ void k_bn_fin(float* sums, const float* __restrict__ g, const float* __restrict__ b,
                         long nrows, int C) {
  int c = threadIdx.x;
  if (c >= C) return;
  float inv = 1.f / (float)nrows;
  float mu = sums[c] * inv;
  float var = sums[C + c] * inv - mu * mu;
  float a = g[c] * rsqrtf(var + EPS);
  sums[c] = a;               // scale
  sums[C + c] = b[c] - mu * a;  // shift
}

// ---------------------------------------------------------------- GAT pieces
__global__ void k_gat_ab(const float* __restrict__ xw, const float* __restrict__ att_s,
                         const float* __restrict__ att_d, float* __restrict__ a_s,
                         float* __restrict__ a_d, long N) {
  long tid = (long)blockIdx.x * blockDim.x + threadIdx.x;
  if (tid >= 2 * N) return;
  long n = tid >> 1; int h = tid & 1;
  const float* xr = xw + (size_t)n * 64 + h * 32;
  const float* as = att_s + h * 32;
  const float* ad = att_d + h * 32;
  float s = 0.f, d = 0.f;
#pragma unroll 8
  for (int c = 0; c < 32; c++) { float v = xr[c]; s += v * as[c]; d += v * ad[c]; }
  a_s[tid] = s; a_d[tid] = d;
}
__global__ void k_gat_m_init(float* m, const float* a_s, const float* a_d, long N2) {
  long i = (long)blockIdx.x * blockDim.x + threadIdx.x;
  if (i < N2) m[i] = lrelu(a_s[i] + a_d[i]);
}
__global__ void k_gat_m_edges(float* m, const float* __restrict__ a_s,
                              const float* __restrict__ a_d, const int* __restrict__ ei, long E) {
  long tid = (long)blockIdx.x * blockDim.x + threadIdx.x;
  if (tid >= 2 * E) return;
  long e = tid >> 1; int h = tid & 1;
  int s = ei[e], d = ei[E + e];
  atomicMaxF(&m[2 * (long)d + h], lrelu(a_s[2 * (long)s + h] + a_d[2 * (long)d + h]));
}
__global__ void k_gat_den_init(float* denom, const float* m, const float* a_s,
                               const float* a_d, long N2) {
  long i = (long)blockIdx.x * blockDim.x + threadIdx.x;
  if (i < N2) denom[i] = __expf(lrelu(a_s[i] + a_d[i]) - m[i]);
}
__global__ void k_gat_den_edges(float* denom, const float* __restrict__ m,
                                const float* __restrict__ a_s, const float* __restrict__ a_d,
                                const int* __restrict__ ei, long E) {
  long tid = (long)blockIdx.x * blockDim.x + threadIdx.x;
  if (tid >= 2 * E) return;
  long e = tid >> 1; int h = tid & 1;
  int s = ei[e], d = ei[E + e];
  long di = 2 * (long)d + h;
  float v = lrelu(a_s[2 * (long)s + h] + a_d[di]);
  atomicAdd(&denom[di], __expf(v - m[di]));
}
__global__ void k_gat_out_init(float* __restrict__ out, const float* __restrict__ xw,
                               const float* __restrict__ a_s, const float* __restrict__ a_d,
                               const float* __restrict__ m, const float* __restrict__ denom,
                               const float* __restrict__ gb, long N) {
  long tid = (long)blockIdx.x * blockDim.x + threadIdx.x;
  if (tid >= N * 64) return;
  long n = tid >> 6; int c = tid & 63; int h = c >> 5;
  long i = 2 * n + h;
  float alpha = __expf(lrelu(a_s[i] + a_d[i]) - m[i]) / (denom[i] + 1e-16f);
  out[tid] = gb[c] + xw[tid] * alpha;
}
__global__ void k_gat_out_edges(float* __restrict__ out, const float* __restrict__ xw,
                                const float* __restrict__ a_s, const float* __restrict__ a_d,
                                const float* __restrict__ m, const float* __restrict__ denom,
                                const int* __restrict__ ei, long E) {
  long tid = (long)blockIdx.x * blockDim.x + threadIdx.x;
  if (tid >= E * 16) return;
  long e = tid >> 4; int g = tid & 15;
  int c0 = g * 4; int h = c0 >> 5;
  int s = ei[e], d = ei[E + e];
  long di = 2 * (long)d + h;
  float v = lrelu(a_s[2 * (long)s + h] + a_d[di]);
  float alpha = __expf(v - m[di]) / (denom[di] + 1e-16f);
  float4 xv = ((const float4*)(xw + (size_t)s * 64))[g];
  float* o = out + (size_t)d * 64 + c0;
  atomicAdd(o + 0, xv.x * alpha); atomicAdd(o + 1, xv.y * alpha);
  atomicAdd(o + 2, xv.z * alpha); atomicAdd(o + 3, xv.w * alpha);
}

// ---------------------------------------------------------------- prediction head
__global__ void __launch_bounds__(256) k_head(const float* __restrict__ h3,
                                              const float* __restrict__ ss,
                                              const float* __restrict__ w1,
                                              const float* __restrict__ b1,
                                              const float* __restrict__ w2,
                                              const float* __restrict__ b2,
                                              float* __restrict__ out, long N) {
  __shared__ float sw1[512], sb1[16], sw2[16];
  int t = threadIdx.x;
  for (int i = t; i < 512; i += 256) sw1[i] = w1[i];
  if (t < 16) { sb1[t] = b1[t]; sw2[t] = w2[t]; }
  __syncthreads();
  long n = (long)blockIdx.x * 256 + t;
  if (n >= N) return;
  float xv[32];
#pragma unroll
  for (int c = 0; c < 32; c++)
    xv[c] = fmaxf(h3[(size_t)n * 32 + c] * ss[c] + ss[32 + c], 0.f);
  float acc = b2[0];
#pragma unroll 4
  for (int j = 0; j < 16; j++) {
    float a = sb1[j];
#pragma unroll
    for (int c = 0; c < 32; c++) a += xv[c] * sw1[c * 16 + j];
    acc += fmaxf(a, 0.f) * sw2[j];
  }
  out[n] = 1.f / (1.f + __expf(-acc));
}

// ---------------------------------------------------------------- launch
extern "C" void kernel_launch(void* const* d_in, const int* in_sizes, int n_in,
                              void* d_out, int out_size, void* d_ws, size_t ws_size,
                              hipStream_t stream) {
  const float* feat  = (const float*)d_in[0];
  const int*   ei    = (const int*)d_in[1];
  const float* ln_g  = (const float*)d_in[2];
  const float* ln_b  = (const float*)d_in[3];
  const float* enc_w1= (const float*)d_in[4];
  const float* enc_b1= (const float*)d_in[5];
  const float* enc_w2= (const float*)d_in[6];
  const float* enc_b2= (const float*)d_in[7];
  const float* gcn1_w= (const float*)d_in[8];
  const float* gcn1_b= (const float*)d_in[9];
  const float* bn1_g = (const float*)d_in[10];
  const float* bn1_b = (const float*)d_in[11];
  const float* gat_w = (const float*)d_in[12];
  const float* gat_as= (const float*)d_in[13];
  const float* gat_ad= (const float*)d_in[14];
  const float* gat_b = (const float*)d_in[15];
  const float* bn2_g = (const float*)d_in[16];
  const float* bn2_b = (const float*)d_in[17];
  const float* gcn2_w= (const float*)d_in[18];
  const float* gcn2_b= (const float*)d_in[19];
  const float* bn3_g = (const float*)d_in[20];
  const float* bn3_b = (const float*)d_in[21];
  const float* pw1   = (const float*)d_in[22];
  const float* pb1   = (const float*)d_in[23];
  const float* pw2   = (const float*)d_in[24];
  const float* pb2   = (const float*)d_in[25];

  const long N = in_sizes[0] / 128;
  const long E = in_sizes[1] / 2;

  size_t off = 0;
  auto alloc = [&](size_t bytes) -> char* {
    char* p = (char*)d_ws + off;
    off += (bytes + 255) & ~(size_t)255;
    return p;
  };
  float* X128  = (float*)alloc((size_t)N * 128 * 4);
  float* H1    = (float*)alloc((size_t)N * 64 * 4);
  float* H2    = (float*)alloc((size_t)N * 64 * 4);
  float* H3    = (float*)alloc((size_t)N * 64 * 4);
  float* dinv  = (float*)alloc((size_t)N * 4);
  float* a_s   = (float*)alloc((size_t)N * 2 * 4);
  float* a_d   = (float*)alloc((size_t)N * 2 * 4);
  float* mbuf  = (float*)alloc((size_t)N * 2 * 4);
  float* denom = (float*)alloc((size_t)N * 2 * 4);
  float* sums1 = (float*)alloc(128 * 4);
  float* sums2 = (float*)alloc(128 * 4);
  float* sums3 = (float*)alloc(128 * 4);
  _Float16* pwE1 = (_Float16*)alloc(128 * 64 * 2);
  _Float16* pwE2 = (_Float16*)alloc(64 * 64 * 2);
  _Float16* pwG1 = (_Float16*)alloc(64 * 64 * 2);
  _Float16* pwGA = (_Float16*)alloc(64 * 64 * 2);
  _Float16* pwG2 = (_Float16*)alloc(64 * 32 * 2);
  (void)ws_size; (void)n_in; (void)out_size;

  auto cdiv = [](long a, long b) { return (unsigned)((a + b - 1) / b); };
  const dim3 B256(256);

  // pack weights to WMMA B-fragment order
  k_pack_w<<<cdiv(128 * 64, 256), B256, 0, stream>>>(enc_w1, pwE1, 128, 64);
  k_pack_w<<<cdiv(64 * 64, 256),  B256, 0, stream>>>(enc_w2, pwE2, 64, 64);
  k_pack_w<<<cdiv(64 * 64, 256),  B256, 0, stream>>>(gcn1_w, pwG1, 64, 64);
  k_pack_w<<<cdiv(64 * 64, 256),  B256, 0, stream>>>(gat_w,  pwGA, 64, 64);
  k_pack_w<<<cdiv(64 * 32, 256),  B256, 0, stream>>>(gcn2_w, pwG2, 64, 32);

  // degree / dinv (self-loop => init 1)
  k_fill<<<cdiv(N, 256), B256, 0, stream>>>(dinv, 1.f, N);
  k_deg_count<<<cdiv(E, 256), B256, 0, stream>>>(dinv, ei, E);
  k_rsqrt<<<cdiv(N, 256), B256, 0, stream>>>(dinv, N);

  // LayerNorm + encoder MLP (WMMA)
  k_layernorm<<<cdiv(N, 8), B256, 0, stream>>>(feat, ln_g, ln_b, X128, N);
  unsigned gB = cdiv(N, 128);
  k_gemm_wmma<<<gB, B256, 128 * 128 * 2, stream>>>(X128, (const unsigned*)pwE1, enc_b1,
      nullptr, 0, 1, H1, N, 128, 64);
  k_gemm_wmma<<<gB, B256, 128 * 64 * 2, stream>>>(H1, (const unsigned*)pwE2, enc_b2,
      nullptr, 0, 1, H2, N, 64, 64);

  // GCN1: xw -> scatter -> H3 (raw)
  k_gemm_wmma<<<gB, B256, 128 * 64 * 2, stream>>>(H2, (const unsigned*)pwG1, nullptr,
      nullptr, 0, 0, H1, N, 64, 64);
  k_gcn_self<<<cdiv(N * 64, 256), B256, 0, stream>>>(H3, H1, dinv, gcn1_b, N, 6);
  k_gcn_edge<<<cdiv(E * 16, 256), B256, 0, stream>>>(H3, H1, dinv, ei, E, 6, 4);

  // BN1 stats -> scale/shift (applied inside next GEMM's A staging, with relu)
  k_fill<<<1, 128, 0, stream>>>(sums1, 0.f, 128);
  k_bn_stats<<<512, B256, 0, stream>>>(H3, sums1, N, 6);
  k_bn_fin<<<1, 64, 0, stream>>>(sums1, bn1_g, bn1_b, N, 64);

  // GAT: xw2 = bnrelu(H3) @ gat_w
  k_gemm_wmma<<<gB, B256, 128 * 64 * 2, stream>>>(H3, (const unsigned*)pwGA, nullptr,
      sums1, 1, 0, H1, N, 64, 64);
  k_gat_ab<<<cdiv(2 * N, 256), B256, 0, stream>>>(H1, gat_as, gat_ad, a_s, a_d, N);
  k_gat_m_init<<<cdiv(2 * N, 256), B256, 0, stream>>>(mbuf, a_s, a_d, 2 * N);
  k_gat_m_edges<<<cdiv(2 * E, 256), B256, 0, stream>>>(mbuf, a_s, a_d, ei, E);
  k_gat_den_init<<<cdiv(2 * N, 256), B256, 0, stream>>>(denom, mbuf, a_s, a_d, 2 * N);
  k_gat_den_edges<<<cdiv(2 * E, 256), B256, 0, stream>>>(denom, mbuf, a_s, a_d, ei, E);
  k_gat_out_init<<<cdiv(N * 64, 256), B256, 0, stream>>>(H2, H1, a_s, a_d, mbuf, denom, gat_b, N);
  k_gat_out_edges<<<cdiv(E * 16, 256), B256, 0, stream>>>(H2, H1, a_s, a_d, mbuf, denom, ei, E);

  // BN2
  k_fill<<<1, 128, 0, stream>>>(sums2, 0.f, 128);
  k_bn_stats<<<512, B256, 0, stream>>>(H2, sums2, N, 6);
  k_bn_fin<<<1, 64, 0, stream>>>(sums2, bn2_g, bn2_b, N, 64);

  // GCN2: xw3 = bnrelu(H2) @ gcn2_w (64->32), scatter -> H3 (N x 32)
  k_gemm_wmma<<<gB, B256, 128 * 64 * 2, stream>>>(H2, (const unsigned*)pwG2, nullptr,
      sums2, 1, 0, H1, N, 64, 32);
  k_gcn_self<<<cdiv(N * 32, 256), B256, 0, stream>>>(H3, H1, dinv, gcn2_b, N, 5);
  k_gcn_edge<<<cdiv(E * 8, 256), B256, 0, stream>>>(H3, H1, dinv, ei, E, 5, 3);

  // BN3 + head
  k_fill<<<1, 128, 0, stream>>>(sums3, 0.f, 128);
  k_bn_stats<<<512, B256, 0, stream>>>(H3, sums3, N, 5);
  k_bn_fin<<<1, 32, 0, stream>>>(sums3, bn3_g, bn3_b, N, 32);
  k_head<<<cdiv(N, 256), B256, 0, stream>>>(H3, sums3, pw1, pb1, pw2, pb2, (float*)d_out, N);
}
